// ConditionEncoder_81046032876084
// MI455X (gfx1250) — matrix operand order repaired and checked by reading, not compiled, not run on hardware
//
#include <hip/hip_runtime.h>
#include <hip/hip_bf16.h>

typedef __attribute__((ext_vector_type(16))) _Float16 v16h;
typedef __attribute__((ext_vector_type(8)))  float    v8f;

#define BDIM   4
#define NNODE  512
#define PPTS   20
#define QDIM   256
#define CDIM   128
#define LSTRIDE 132   // 128 + 4 pad: conflict-free LDS writes from WMMA layout

// ---------------- setup kernels (tiny; feed the main pair kernel) -------------

// Per-node geometry: start, end, normalized direction vector.
__global__ void k_node(const float* __restrict__ pts, float* __restrict__ nd) {
    int t = blockIdx.x * blockDim.x + threadIdx.x;           // 0..2047  (b*512+n)
    if (t >= BDIM * NNODE) return;
    const float* p = pts + (size_t)t * (PPTS * 2);
    float sx = p[0], sy = p[1];
    float ex = p[2 * (PPTS - 1)], ey = p[2 * (PPTS - 1) + 1];
    float dx = ex - sx, dy = ey - sy;
    float inv = 1.0f / (sqrtf(dx * dx + dy * dy) + 1e-6f);
    float* o = nd + (size_t)t * 6;
    o[0] = sx; o[1] = sy; o[2] = ex; o[3] = ey; o[4] = dx * inv; o[5] = dy * inv;
}

// t1 = relu(query_feat @ wq1 + bq1) : (2048 x 128)
__global__ void k_t1(const float* __restrict__ qf, const float* __restrict__ wq1,
                     const float* __restrict__ bq1, float* __restrict__ t1) {
    int t = blockIdx.x * blockDim.x + threadIdx.x;           // 2048*128
    int r = t >> 7, m = t & 127;
    const float* x = qf + (size_t)r * QDIM;
    float s = bq1[m];
    for (int k = 0; k < QDIM; ++k) s += x[k] * wq1[k * CDIM + m];
    t1[t] = fmaxf(s, 0.0f);
}

// q = t1 @ wq2 + bq2 : (2048 x 64)
__global__ void k_q(const float* __restrict__ t1, const float* __restrict__ wq2,
                    const float* __restrict__ bq2, float* __restrict__ q) {
    int t = blockIdx.x * blockDim.x + threadIdx.x;           // 2048*64
    int r = t >> 6, j = t & 63;
    const float* x = t1 + (size_t)r * CDIM;
    float s = bq2[j];
    for (int m = 0; m < CDIM; ++m) s += x[m] * wq2[m * 64 + j];
    q[t] = s;
}

// hA[r, (c&15)*8 + (c>>4)] = q[r,0:64]@wf[0:64,c] + bf[c] + bg2@wf[128:160,c]
//   (channel-permuted so the hot kernel reads its 8 tile values contiguously)
// hB[r,c] = q[r,0:64]@wf[64:128,c]   (natural order; consumed contiguously)
__global__ void k_hab(const float* __restrict__ q, const float* __restrict__ wf,
                      const float* __restrict__ bf, const float* __restrict__ bg2,
                      float* __restrict__ hA, float* __restrict__ hB) {
    int t = blockIdx.x * blockDim.x + threadIdx.x;           // 2048*128
    int r = t >> 7, c = t & 127;
    const float* x = q + (size_t)r * 64;
    float sa = bf[c], sb = 0.0f;
    for (int k = 0; k < 64; ++k) {
        sa += x[k] * wf[k * CDIM + c];
        sb += x[k] * wf[(64 + k) * CDIM + c];
    }
    float cb = 0.0f;
    for (int m = 0; m < 32; ++m) cb += bg2[m] * wf[(128 + m) * CDIM + c];
    hA[(size_t)r * CDIM + (c & 15) * 8 + (c >> 4)] = sa + cb;
    hB[t] = sb;
}

// W = wg2 @ wf[128:160,:]  (32 x 128), pre-swizzled into wave32 WMMA f16
// B-matrix layout: tile t (cols 16t..16t+15); lanes 0-15: K=0..7,16..23, lanes 16-31: K=8..15,24..31.
__global__ void k_wh(const float* __restrict__ wg2, const float* __restrict__ wf,
                     _Float16* __restrict__ Wh) {
    int idx = blockIdx.x * blockDim.x + threadIdx.x;         // 8*32*16 = 4096
    int t = idx >> 9, l = (idx >> 4) & 31, e = idx & 15;
    int K = ((l >= 16) ? 8 : 0) + e + ((e >= 8) ? 8 : 0);
    int c = t * 16 + (l & 15);
    float s = 0.0f;
    for (int m = 0; m < 32; ++m) s += wg2[K * 32 + m] * wf[(128 + m) * CDIM + c];
    Wh[idx] = (_Float16)s;
}

// wg1/bg1 pre-swizzled per-lane SoA: out[l*80 + m*16 + e], m = {w0,w1,w2,w3,b}
// for the lane's 16 A-matrix K-slots (ISA 16-bit A 16x32 layout).
__global__ void k_wg1sw(const float* __restrict__ wg1, const float* __restrict__ bg1,
                        float* __restrict__ o) {
    int idx = blockIdx.x * blockDim.x + threadIdx.x;         // 32*16 = 512
    if (idx >= 512) return;
    int l = idx >> 4, e = idx & 15;
    int k = ((l >= 16) ? 8 : 0) + e + ((e >= 8) ? 8 : 0);
    float* p = o + l * 80;
    p[e]      = wg1[k];
    p[16 + e] = wg1[32 + k];
    p[32 + e] = wg1[64 + k];
    p[48 + e] = wg1[96 + k];
    p[64 + e] = bg1[k];
}

// ---------------- main pair kernel -------------------------------------------
// One wave per (b, i, 16-wide j tile): geo MLP layer1 -> f16 A (16x32),
// 8x v_wmma_f32_16x16x32_f16 against pre-swizzled W, +hA, LDS transpose,
// +hB, layernorm(128) with one xor-16 shuffle, relu, float4 stores.
__global__ void __launch_bounds__(128)
k_pair(const float* __restrict__ nd, const float* __restrict__ hA,
       const float* __restrict__ hB, const _Float16* __restrict__ Wh,
       const float* __restrict__ wg1sw, const float* __restrict__ lnw,
       const float* __restrict__ lnb, float* __restrict__ out) {
    __shared__ float lds[4 * 16 * LSTRIDE];

    const int lane = threadIdx.x & 31;
    const int wv   = threadIdx.x >> 5;
    const int u    = blockIdx.x * 4 + wv;      // 0..65535
    const int jt   = u & 31;
    const int i    = (u >> 5) & (NNODE - 1);
    const int b    = u >> 14;
    const int j0   = jt << 4;
    const int rlo  = lane & 15;                // j-local / N index
    const int hsel = lane >> 4;

    // geometry for pair (i, j0+rlo); i-side terms are wave-constant
    const float* ndB = nd + (size_t)b * NNODE * 6;
    const float exi = ndB[i * 6 + 2], eyi = ndB[i * 6 + 3];
    const float dvx = ndB[i * 6 + 4], dvy = ndB[i * 6 + 5];
    const int   j   = j0 + rlo;
    const float cx  = ndB[j * 6 + 0] - exi;
    const float cy  = ndB[j * 6 + 1] - eyi;
    const float dist = sqrtf(cx * cx + cy * cy);
    const float inv  = 1.0f / (dist + 1e-6f);
    const float algn = (dvx * cx + dvy * cy) * inv;

    // per-lane swizzled geo-MLP weights: 20 aligned b128 loads
    const float* wp = wg1sw + lane * 80;
    float w0[16], w1[16], w2[16], w3[16], bb[16];
    #pragma unroll
    for (int r4 = 0; r4 < 4; ++r4) {
        *(float4*)&w0[r4 * 4] = *(const float4*)(wp +      r4 * 4);
        *(float4*)&w1[r4 * 4] = *(const float4*)(wp + 16 + r4 * 4);
        *(float4*)&w2[r4 * 4] = *(const float4*)(wp + 32 + r4 * 4);
        *(float4*)&w3[r4 * 4] = *(const float4*)(wp + 48 + r4 * 4);
        *(float4*)&bb[r4 * 4] = *(const float4*)(wp + 64 + r4 * 4);
    }

    // A matrix: r_k = relu(geo @ wg1 + bg1), 16 K-slots per lane
    v16h a;
    #pragma unroll
    for (int e = 0; e < 16; ++e) {
        float r = dist * w0[e] + algn * w1[e] + cx * w2[e] + cy * w3[e] + bb[e];
        a[e] = (_Float16)fmaxf(r, 0.0f);
    }

    // 8 WMMAs: D(16x16) per channel-tile; W pre-swizzled -> contiguous v16h
    v8f acc[8];
    const _Float16* wl = Wh + lane * 16;
    #pragma unroll
    for (int t = 0; t < 8; ++t) {
        v16h bm = *(const v16h*)(wl + t * 512);
        v8f  z  = {};
        acc[t] = __builtin_amdgcn_wmma_f32_16x16x32_f16(
            false, a, false, bm, (short)0, z, false, false);
    }

    // channel-only term hA[i]: permuted layout -> 2 b128 loads
    const float* hAp = hA + ((size_t)(b * NNODE + i)) * CDIM + rlo * 8;
    float hv[8];
    *(float4*)&hv[0] = *(const float4*)(hAp);
    *(float4*)&hv[4] = *(const float4*)(hAp + 4);

    // spill D tiles (+hA) to LDS (conflict-free via stride 132)
    float* ldsw = lds + wv * 16 * LSTRIDE;
    #pragma unroll
    for (int t = 0; t < 8; ++t) {
        int cch = t * 16 + rlo;
        #pragma unroll
        for (int m = 0; m < 8; ++m) {
            int r2 = m + hsel * 8;
            ldsw[r2 * LSTRIDE + cch] = acc[t][m] + hv[t];
        }
    }
    __syncthreads();

    // phase 2: lane handles row (j0+rlo), 64 channels (hsel half)
    const float* ldsr = ldsw + rlo * LSTRIDE + hsel * 64;
    const float* hBp  = hB + ((size_t)(b * NNODE) + j0 + rlo) * CDIM + hsel * 64;
    float v[64];
    float s = 0.0f, sq = 0.0f;
    #pragma unroll
    for (int c4 = 0; c4 < 16; ++c4) {
        float4 lv = *(const float4*)(ldsr + c4 * 4);
        float4 hvv = *(const float4*)(hBp + c4 * 4);
        float x0 = lv.x + hvv.x, x1 = lv.y + hvv.y, x2 = lv.z + hvv.z, x3 = lv.w + hvv.w;
        v[c4 * 4 + 0] = x0; v[c4 * 4 + 1] = x1; v[c4 * 4 + 2] = x2; v[c4 * 4 + 3] = x3;
        s  += x0 + x1 + x2 + x3;
        sq += x0 * x0 + x1 * x1 + x2 * x2 + x3 * x3;
    }
    float st  = s  + __shfl_xor(s,  16, 32);
    float sqt = sq + __shfl_xor(sq, 16, 32);
    float mean = st * (1.0f / 128.0f);
    float var  = sqt * (1.0f / 128.0f) - mean * mean;
    float rstd = rsqrtf(var + 1e-5f);

    float* op = out + (((size_t)(b * NNODE + i)) * NNODE + j0 + rlo) * CDIM + hsel * 64;
    const float* lw = lnw + hsel * 64;
    const float* lb = lnb + hsel * 64;
    #pragma unroll
    for (int c4 = 0; c4 < 16; ++c4) {
        float4 o;
        o.x = fmaxf((v[c4 * 4 + 0] - mean) * rstd * lw[c4 * 4 + 0] + lb[c4 * 4 + 0], 0.0f);
        o.y = fmaxf((v[c4 * 4 + 1] - mean) * rstd * lw[c4 * 4 + 1] + lb[c4 * 4 + 1], 0.0f);
        o.z = fmaxf((v[c4 * 4 + 2] - mean) * rstd * lw[c4 * 4 + 2] + lb[c4 * 4 + 2], 0.0f);
        o.w = fmaxf((v[c4 * 4 + 3] - mean) * rstd * lw[c4 * 4 + 3] + lb[c4 * 4 + 3], 0.0f);
        *(float4*)(op + c4 * 4) = o;
    }
}

// ---------------- launcher ---------------------------------------------------

extern "C" void kernel_launch(void* const* d_in, const int* in_sizes, int n_in,
                              void* d_out, int out_size, void* d_ws, size_t ws_size,
                              hipStream_t stream) {
    const float* qf  = (const float*)d_in[0];
    const float* pts = (const float*)d_in[1];
    const float* wq1 = (const float*)d_in[2];
    const float* bq1 = (const float*)d_in[3];
    const float* wq2 = (const float*)d_in[4];
    const float* bq2 = (const float*)d_in[5];
    const float* wg1 = (const float*)d_in[6];
    const float* bg1 = (const float*)d_in[7];
    const float* wg2 = (const float*)d_in[8];
    const float* bg2 = (const float*)d_in[9];
    const float* wf  = (const float*)d_in[10];
    const float* bf  = (const float*)d_in[11];
    const float* lnw = (const float*)d_in[12];
    const float* lnb = (const float*)d_in[13];
    float* out = (float*)d_out;

    // workspace layout (float offsets)
    float* ws = (float*)d_ws;
    float* nd = ws;                       //  2048*6   = 12288
    float* t1 = ws + 12288;               //  2048*128 = 262144
    float* q  = ws + 274432;              //  2048*64  = 131072
    float* hA = ws + 405504;              //  2048*128 (channel-permuted)
    float* hB = ws + 667648;              //  2048*128
    _Float16* Wh = (_Float16*)(ws + 929792);  // 4096 halves (32B-aligned)
    float* wg1sw = ws + 931840;           //  32*80 = 2560

    k_node <<<8,    256, 0, stream>>>(pts, nd);
    k_t1   <<<1024, 256, 0, stream>>>(qf, wq1, bq1, t1);
    k_q    <<<512,  256, 0, stream>>>(t1, wq2, bq2, q);
    k_hab  <<<1024, 256, 0, stream>>>(q, wf, bf, bg2, hA, hB);
    k_wh   <<<16,   256, 0, stream>>>(wg2, wf, Wh);
    k_wg1sw<<<2,    256, 0, stream>>>(wg1, bg1, wg1sw);

    // 4*512*32 = 65536 wave-units, 4 waves/block
    k_pair<<<16384, 128, 0, stream>>>(nd, hA, hB, Wh, wg1sw, lnw, lnb, out);
}
